// VectorQuantization_2559800508595
// MI455X (gfx1250) — matrix operand order repaired and checked by reading, not compiled, not run on hardware
//
#include <hip/hip_runtime.h>
#include <hip/hip_bf16.h>
#include <cstdint>

typedef __attribute__((ext_vector_type(16))) __bf16 v16bf;
typedef __attribute__((ext_vector_type(8)))  float  v8f;

#define N_ROWS      262144
#define DIM         64
#define KCODES      1024
#define CHUNK       128     // codes staged in LDS per iteration
#define ROWS_PER_WG 256     // 8 waves x 32 rows (2 C-tiles per wave)

// ---------------- CDNA5 async-to-LDS (guarded; falls back to sync copy) ----
#if defined(__has_builtin)
#  if __has_builtin(__builtin_amdgcn_global_load_async_to_lds_b128)
#    define HAVE_ASYNC_B128 1
#  endif
#  if __has_builtin(__builtin_amdgcn_s_wait_asynccnt)
#    define HAVE_WAIT_ASYNC 1
#  endif
#endif

typedef int v4i_b __attribute__((vector_size(16)));
typedef __attribute__((address_space(1))) v4i_b* as1_v4i;
typedef __attribute__((address_space(3))) v4i_b* as3_v4i;

__device__ __forceinline__ void copy16_to_lds(const void* g, void* l) {
#if defined(HAVE_ASYNC_B128)
    __builtin_amdgcn_global_load_async_to_lds_b128((as1_v4i)g, (as3_v4i)l, 0, 0);
#else
    *(uint4*)l = *(const uint4*)g;
#endif
}
__device__ __forceinline__ void wait_lds_copies() {
#if defined(HAVE_ASYNC_B128)
#  if defined(HAVE_WAIT_ASYNC)
    __builtin_amdgcn_s_wait_asynccnt(0);
#  else
    asm volatile("s_wait_asynccnt 0" ::: "memory");
#  endif
#endif
}

// round-to-nearest-even f32 -> bf16 bits
__device__ __forceinline__ unsigned short bf16_rne_bits(float f) {
    uint32_t u = __builtin_bit_cast(uint32_t, f);
    u += 0x7FFFu + ((u >> 16) & 1u);
    return (unsigned short)(u >> 16);
}
__device__ __forceinline__ float bf16_bits_to_f32(unsigned short h) {
    uint32_t u = ((uint32_t)h) << 16;
    return __builtin_bit_cast(float, u);
}

// ---------------- one-shot codebook conversion: f32 -> bf16 hi/lo + |w|^2 --
__global__ void vq_prep_kernel(const float* __restrict__ w,
                               unsigned short* __restrict__ wh,
                               unsigned short* __restrict__ wl,
                               float* __restrict__ wsq_g) {
    const int c = blockIdx.x * blockDim.x + threadIdx.x;   // one code row
    if (c >= KCODES) return;
    const float* wr = w + (size_t)c * DIM;
    float ssq = 0.f;
    #pragma unroll
    for (int d = 0; d < DIM; ++d) {
        float v = wr[d];
        ssq += v * v;
        unsigned short hb = bf16_rne_bits(v);
        wh[(size_t)c * DIM + d] = hb;
        wl[(size_t)c * DIM + d] = bf16_rne_bits(v - bf16_bits_to_f32(hb));
    }
    wsq_g[c] = ssq;
}

// ---------------- main kernel -------------------------------------------
template <bool PRE>
__global__ __launch_bounds__(256) void vq_argmin_wmma_kernel(
    const float* __restrict__ x,           // [N, 64]
    const float* __restrict__ w,           // [1024, 64] (f32, for gather + fallback)
    const unsigned short* __restrict__ wh, // [1024, 64] bf16-hi (PRE)
    const unsigned short* __restrict__ wl, // [1024, 64] bf16-lo (PRE)
    const float* __restrict__ wsq_g,       // [1024] |w|^2 (PRE)
    float* __restrict__ outq,              // [N, 64]
    float* __restrict__ outidx)            // [N] (indices stored as float)
{
    __shared__ alignas(32) unsigned short ws_hi[CHUNK * DIM];   // 16 KB
    __shared__ alignas(32) unsigned short ws_lo[CHUNK * DIM];   // 16 KB
    __shared__ float wsq[CHUNK];                                // 0.5 KB

    const int tid  = threadIdx.x;
    const int wave = tid >> 5;
    const int lane = tid & 31;
    const int col  = lane & 15;   // N-column of B/C fragment
    const int half = lane >> 4;   // K-subrange (A/B) and M-offset (C) selector

    const int rowbase = blockIdx.x * ROWS_PER_WG + wave * 32;

    // ---- A fragments for 2 row-tiles (x as bf16 hi+lo), 16-bit A 16x32 in this code's contiguous k order (B matches):
    // lanes 0-15:  elems 0-15 -> K=0..15
    // lanes 16-31: elems 0-15 -> K=16..31
    v16bf a_hi[2][2], a_lo[2][2];          // [row-tile][k-step]
    #pragma unroll
    for (int rt = 0; rt < 2; ++rt) {
        const float* xr = x + (size_t)(rowbase + rt * 16 + col) * DIM;
        #pragma unroll
        for (int s = 0; s < 2; ++s) {
            #pragma unroll
            for (int run = 0; run < 2; ++run) {
                const int dbase = s * 32 + 16 * half + 8 * run;
                #pragma unroll
                for (int j = 0; j < 8; ++j) {
                    float v = xr[dbase + j];
                    unsigned short hb = bf16_rne_bits(v);
                    a_hi[rt][s][run * 8 + j] = __builtin_bit_cast(__bf16, hb);
                    a_lo[rt][s][run * 8 + j] =
                        __builtin_bit_cast(__bf16, bf16_rne_bits(v - bf16_bits_to_f32(hb)));
                }
            }
        }
    }

    float minval[2][8];
    int   minidx[2][8];
    #pragma unroll
    for (int rt = 0; rt < 2; ++rt)
        #pragma unroll
        for (int r = 0; r < 8; ++r) { minval[rt][r] = 3.4e38f; minidx[rt][r] = 0; }

    for (int chunk = 0; chunk < KCODES / CHUNK; ++chunk) {
        const int cbase = chunk * CHUNK;
        __syncthreads();  // previous chunk's compute done before overwrite

        if (PRE) {
            // ---- pure copy staging: 64B/thread per array, async-to-LDS
            const unsigned short* gh = wh + (size_t)cbase * DIM + tid * 32;
            const unsigned short* gl = wl + (size_t)cbase * DIM + tid * 32;
            #pragma unroll
            for (int q = 0; q < 4; ++q) {
                copy16_to_lds(gh + q * 8, &ws_hi[tid * 32 + q * 8]);
                copy16_to_lds(gl + q * 8, &ws_lo[tid * 32 + q * 8]);
            }
            if (tid < CHUNK) wsq[tid] = wsq_g[cbase + tid];
            wait_lds_copies();
        } else {
            // ---- fallback: convert inline (each thread: half a code row)
            const int code_l = tid >> 1;
            const int dh     = (tid & 1) * 32;
            const float4* wp4 =
                reinterpret_cast<const float4*>(w + (size_t)(cbase + code_l) * DIM + dh);
            float ssq = 0.f;
            #pragma unroll
            for (int q = 0; q < 8; ++q) {
                float4 v4 = wp4[q];
                float vv[4] = {v4.x, v4.y, v4.z, v4.w};
                #pragma unroll
                for (int j = 0; j < 4; j += 2) {
                    float v0 = vv[j], v1 = vv[j + 1];
                    ssq += v0 * v0 + v1 * v1;
                    unsigned short h0 = bf16_rne_bits(v0);
                    unsigned short h1 = bf16_rne_bits(v1);
                    unsigned short l0 = bf16_rne_bits(v0 - bf16_bits_to_f32(h0));
                    unsigned short l1 = bf16_rne_bits(v1 - bf16_bits_to_f32(h1));
                    const int off = code_l * DIM + dh + q * 4 + j;
                    *reinterpret_cast<uint32_t*>(&ws_hi[off]) =
                        (uint32_t)h0 | ((uint32_t)h1 << 16);
                    *reinterpret_cast<uint32_t*>(&ws_lo[off]) =
                        (uint32_t)l0 | ((uint32_t)l1 << 16);
                }
            }
            float other = __shfl_xor(ssq, 1, 32);
            if ((tid & 1) == 0) wsq[code_l] = ssq + other;
        }
        __syncthreads();

        if (PRE && chunk + 1 < KCODES / CHUNK) {
            __builtin_prefetch(wh + (size_t)(cbase + CHUNK) * DIM + tid * 32, 0, 1);
        }

        // ---- 8 code tiles of 16; 4 B-fragment loads feed 12 WMMAs (2 C-tiles)
        #pragma unroll
        for (int tile = 0; tile < CHUNK / 16; ++tile) {
            const int bcode = tile * 16 + col;
            // B 32x16 in the same k order as A: lane(n=col, half) element e -> K = 16*half + e
            // => each lane reads 16 consecutive d of one code row (32B aligned)
            const unsigned short* base_h = &ws_hi[bcode * DIM + 16 * half];
            const unsigned short* base_l = &ws_lo[bcode * DIM + 16 * half];
            v16bf b_hi0 = *reinterpret_cast<const v16bf*>(base_h);        // k-step 0
            v16bf b_hi1 = *reinterpret_cast<const v16bf*>(base_h + 32);   // k-step 1
            v16bf b_lo0 = *reinterpret_cast<const v16bf*>(base_l);
            v16bf b_lo1 = *reinterpret_cast<const v16bf*>(base_l + 32);

            const float wq    = wsq[tile * 16 + col];
            const int   cglob = cbase + tile * 16 + col;

            #pragma unroll
            for (int rt = 0; rt < 2; ++rt) {
                // Two independent accumulator chains (3 WMMAs each) per C-tile;
                // dropped lo*lo term (~2^-18 rel -> f32-grade argmin accuracy).
                v8f accA = {}, accB = {};
                accA = __builtin_amdgcn_wmma_f32_16x16x32_bf16(false, a_hi[rt][0], false, b_hi0, (short)0, accA, false, false);
                accB = __builtin_amdgcn_wmma_f32_16x16x32_bf16(false, a_hi[rt][0], false, b_lo0, (short)0, accB, false, false);
                accA = __builtin_amdgcn_wmma_f32_16x16x32_bf16(false, a_hi[rt][1], false, b_hi1, (short)0, accA, false, false);
                accB = __builtin_amdgcn_wmma_f32_16x16x32_bf16(false, a_hi[rt][1], false, b_lo1, (short)0, accB, false, false);
                accA = __builtin_amdgcn_wmma_f32_16x16x32_bf16(false, a_lo[rt][0], false, b_hi0, (short)0, accA, false, false);
                accB = __builtin_amdgcn_wmma_f32_16x16x32_bf16(false, a_lo[rt][1], false, b_hi1, (short)0, accB, false, false);

                #pragma unroll
                for (int r = 0; r < 8; ++r) {
                    // dist - |x|^2 = |w|^2 - 2 x.w  (|x|^2 doesn't affect argmin)
                    float val = __builtin_fmaf(-2.0f, accA[r] + accB[r], wq);
                    if (val < minval[rt][r]) { minval[rt][r] = val; minidx[rt][r] = cglob; }
                }
            }
        }
    }

    // ---- Min+argmin reduce across the 16 columns (stays within half-wave)
    #pragma unroll
    for (int m = 1; m < 16; m <<= 1) {
        #pragma unroll
        for (int rt = 0; rt < 2; ++rt) {
            #pragma unroll
            for (int r = 0; r < 8; ++r) {
                float ov = __shfl_xor(minval[rt][r], m, 32);
                int   oi = __shfl_xor(minidx[rt][r], m, 32);
                if (ov < minval[rt][r] ||
                    (ov == minval[rt][r] && oi < minidx[rt][r])) {
                    minval[rt][r] = ov; minidx[rt][r] = oi;
                }
            }
        }
    }

    // ---- Output: C layout -> VGPR r holds row (rt*16 + r + 8*half); all 16
    // lanes of a half-group know the argmin -> cooperative float4 gather.
    #pragma unroll
    for (int rt = 0; rt < 2; ++rt) {
        #pragma unroll
        for (int r = 0; r < 8; ++r) {
            const int row = rowbase + rt * 16 + r + 8 * half;
            const int idx = minidx[rt][r];
            const float4 src =
                *reinterpret_cast<const float4*>(w + (size_t)idx * DIM + col * 4);
            *reinterpret_cast<float4*>(outq + (size_t)row * DIM + col * 4) = src;
            if (col == 0) outidx[row] = (float)idx;
        }
    }
}

extern "C" void kernel_launch(void* const* d_in, const int* in_sizes, int n_in,
                              void* d_out, int out_size, void* d_ws, size_t ws_size,
                              hipStream_t stream) {
    const float* x = (const float*)d_in[0];   // [N, 64] f32
    const float* w = (const float*)d_in[1];   // [1024, 64] f32
    float* outq   = (float*)d_out;                      // quantized [N,64]
    float* outidx = outq + (size_t)N_ROWS * DIM;        // indices [N] as float

    const size_t need = (size_t)KCODES * DIM * 2 * sizeof(unsigned short)
                      + (size_t)KCODES * sizeof(float);

    dim3 grid(N_ROWS / ROWS_PER_WG);   // 1024
    dim3 block(256);                   // 8 waves (wave32)

    if (ws_size >= need) {
        unsigned short* wh = (unsigned short*)d_ws;
        unsigned short* wl = wh + (size_t)KCODES * DIM;
        float* wsq_g = (float*)(wl + (size_t)KCODES * DIM);
        vq_prep_kernel<<<dim3(16), dim3(64), 0, stream>>>(w, wh, wl, wsq_g);
        vq_argmin_wmma_kernel<true><<<grid, block, 0, stream>>>(
            x, w, wh, wl, wsq_g, outq, outidx);
    } else {
        vq_argmin_wmma_kernel<false><<<grid, block, 0, stream>>>(
            x, w, nullptr, nullptr, nullptr, outq, outidx);
    }
}
